// Emb_Split_Model_89996744720812
// MI455X (gfx1250) — matrix operand order repaired and checked by reading, not compiled, not run on hardware
//
#include <hip/hip_runtime.h>
#include <hip/hip_fp16.h>

typedef __attribute__((ext_vector_type(16))) _Float16 v16h;
typedef __attribute__((ext_vector_type(8)))  float    v8f;

#define NDRUG 4000
#define NPROT 19000
#define NCELL 1000
#define HD    128
#define NLAYER 2
#define NBATCH 30000
#define EDP 200000
#define EPP 300000
#define ECP 150000
#define MLP_CHUNK 8192

// ---------------------------------------------------------------------------
// Weight convert+transpose: Bt[n*K + k] = (f16) B[k*N + n]
// ---------------------------------------------------------------------------
__global__ void cvt_transpose_f16(const float* __restrict__ B,
                                  _Float16* __restrict__ Bt, int K, int N)
{
    const int i = blockIdx.x * blockDim.x + threadIdx.x;   // write-contiguous
    if (i >= K * N) return;
    const int n = i / K, k = i % K;
    Bt[(size_t)n * K + k] = (_Float16)B[(size_t)k * N + n];
}

// ---------------------------------------------------------------------------
// WMMA GEMM: C[M,N] = optrelu(A[M,K] @ B[K,N] + bias[N])
// A fp32 row-major; B pre-converted f16, transposed layout Bt[N][K].
// One wave computes a 16x128 output tile (8 WMMAs / k-step, A frag reused 8x).
// All 8 B fragments are loaded into distinct registers up front so the
// compiler can issue one load clause and use progressive partial
// s_wait_loadcnt levels across the WMMA chain (instead of a full drain
// before every WMMA).
// Requires N % 128 == 0, K % 32 == 0 (true for all GEMMs here).
// M-tail rows are CLAMPED (not masked) so all loads are unconditional;
// clamped rows only affect output rows that are never stored.
// ---------------------------------------------------------------------------
__global__ void __launch_bounds__(128)
gemm_wmma_f16(const float* __restrict__ A, const _Float16* __restrict__ Bt,
              const float* __restrict__ bias, float* __restrict__ C,
              int M, int N, int K, int dorelu)
{
    const int wid  = threadIdx.x >> 5;
    const int lane = threadIdx.x & 31;
    const int ntn  = N >> 7;                     // 128-column groups
    const int ntiles = ntn * ((M + 15) >> 4);
    const int tile = blockIdx.x * 4 + wid;
    if (tile >= ntiles) return;                  // wave-uniform exit
    const int tn = tile % ntn;
    const int tm = tile / ntn;

    const int hi  = lane >> 4;                   // half-wave id
    const int l16 = lane & 15;
    int row = tm * 16 + l16;
    if (row >= M) row = M - 1;                   // clamp tail (see above)
    const float4* Arow4 = (const float4*)(A + (size_t)row * K);
    const int col0 = tn * 128 + l16;

    // per-fragment B base pointers (hoisted out of the k-loop)
    const _Float16* bbase[8];
    #pragma unroll
    for (int j = 0; j < 8; ++j)
        bbase[j] = Bt + (size_t)(col0 + 16 * j) * K + hi * 16;

    v8f acc[8] = {v8f{}, v8f{}, v8f{}, v8f{}, v8f{}, v8f{}, v8f{}, v8f{}};

    for (int kb = 0; kb < K; kb += 32) {
        // ---- A fragment (16x32 f16), loads issued first ----
        // lanes 0-15: K {kb+0..15}; lanes 16-31: K {kb+16..31}
        v16h af;
        {
            const int i0 = (kb + hi * 16) >> 2;  // float4 index
            const float4 a0 = Arow4[i0];
            const float4 a1 = Arow4[i0 + 1];
            const float4 a2 = Arow4[i0 + 2];
            const float4 a3 = Arow4[i0 + 3];
            af[0]=(_Float16)a0.x;  af[1]=(_Float16)a0.y;
            af[2]=(_Float16)a0.z;  af[3]=(_Float16)a0.w;
            af[4]=(_Float16)a1.x;  af[5]=(_Float16)a1.y;
            af[6]=(_Float16)a1.z;  af[7]=(_Float16)a1.w;
            af[8]=(_Float16)a2.x;  af[9]=(_Float16)a2.y;
            af[10]=(_Float16)a2.z; af[11]=(_Float16)a2.w;
            af[12]=(_Float16)a3.x; af[13]=(_Float16)a3.y;
            af[14]=(_Float16)a3.z; af[15]=(_Float16)a3.w;
        }
        if (kb + 32 < K)                         // prefetch next A k-block
            __builtin_prefetch((const void*)(Arow4 + (((kb + 32) + hi * 16) >> 2)), 0, 3);

        // ---- all 8 B fragments into distinct registers (one load clause) ----
        v16h bf[8];
        #pragma unroll
        for (int j = 0; j < 8; ++j)
            bf[j] = *(const v16h*)(bbase[j] + kb);

        // ---- WMMA chain; waits drain progressively as fragments arrive ----
        #pragma unroll
        for (int j = 0; j < 8; ++j)
            acc[j] = __builtin_amdgcn_wmma_f32_16x16x32_f16(
                         false, af, false, bf[j], (short)0, acc[j], false, false);
    }

    #pragma unroll
    for (int r = 0; r < 8; ++r) {
        const int m = tm * 16 + r + hi * 8;
        if (m < M) {
            float* crow = C + (size_t)m * N;
            #pragma unroll
            for (int j = 0; j < 8; ++j) {
                const int col = col0 + 16 * j;
                float v = acc[j][r] + (bias ? bias[col] : 0.0f);
                if (dorelu) v = fmaxf(v, 0.0f);
                crow[col] = v;
            }
        }
    }
}

// ---------------------------------------------------------------------------
// wa = W @ a  for both attention vectors ((x@W).a == x.(W@a))
// ---------------------------------------------------------------------------
__global__ void wa_kernel(const float* __restrict__ W,
                          const float* __restrict__ a_s,
                          const float* __restrict__ a_d,
                          float* __restrict__ wa_s, float* __restrict__ wa_d)
{
    __shared__ float sh[2][HD];
    const int t = threadIdx.x;                  // 256 threads
    if (t < HD) sh[0][t] = a_s[t];
    else        sh[1][t - HD] = a_d[t - HD];
    __syncthreads();
    const int which = t >> 7;
    const int rowi  = t & (HD - 1);
    const float* wr = W + (size_t)rowi * HD;
    const float* av = sh[which];
    float s = 0.f;
    #pragma unroll 8
    for (int j = 0; j < HD; ++j) s += wr[j] * av[j];
    (which ? wa_d : wa_s)[rowi] = s;
}

// per-node dot: out[n] = X[n,:] . v   (one wave per node)
__global__ void rowdot_kernel(const float* __restrict__ X,
                              const float* __restrict__ v,
                              float* __restrict__ out, int Nn)
{
    const int node = blockIdx.x * (blockDim.x >> 5) + (threadIdx.x >> 5);
    if (node >= Nn) return;
    const int lane = threadIdx.x & 31;
    const float* xr = X + (size_t)node * HD;
    float s = 0.f;
    #pragma unroll
    for (int i = 0; i < 4; ++i) s += xr[lane + 32 * i] * v[lane + 32 * i];
    #pragma unroll
    for (int off = 16; off > 0; off >>= 1) s += __shfl_down(s, off, 32);
    if (lane == 0) out[node] = s;
}

// ---------------------------------------------------------------------------
// segment softmax over dst + weighted aggregation (3 edge passes, atomics)
// ---------------------------------------------------------------------------
__device__ __forceinline__ unsigned encf(float f) {
    unsigned u = __float_as_uint(f);
    return (u & 0x80000000u) ? ~u : (u | 0x80000000u);
}
__device__ __forceinline__ float decf(unsigned u) {
    return __uint_as_float((u & 0x80000000u) ? (u & 0x7fffffffu) : ~u);
}

__global__ void init_attn_kernel(unsigned* __restrict__ m_enc,
                                 float* __restrict__ denom, int Nd)
{
    const int i = blockIdx.x * blockDim.x + threadIdx.x;
    if (i < Nd) { m_enc[i] = 0x00800000u /* enc(-FLT_MAX) */; denom[i] = 0.f; }
}

__global__ void zero_kernel(float* __restrict__ p, int n)
{
    const int i = blockIdx.x * blockDim.x + threadIdx.x;
    if (i < n) p[i] = 0.f;
}

__global__ void edge_max_kernel(const int* __restrict__ src, const int* __restrict__ dst,
                                int E, int nloops,
                                const float* __restrict__ ssrc,
                                const float* __restrict__ sdstv,
                                unsigned* __restrict__ m_enc)
{
    const int e = blockIdx.x * blockDim.x + threadIdx.x;
    if (e >= E + nloops) return;
    int s_, d_;
    if (e < E) { s_ = src[e]; d_ = dst[e]; } else { s_ = d_ = e - E; }
    const float x = ssrc[s_] + sdstv[d_];
    const float ee = x >= 0.f ? x : 0.2f * x;     // leaky_relu(0.2)
    atomicMax(&m_enc[d_], encf(ee));
}

__global__ void edge_expsum_kernel(const int* __restrict__ src, const int* __restrict__ dst,
                                   int E, int nloops,
                                   const float* __restrict__ ssrc,
                                   const float* __restrict__ sdstv,
                                   const unsigned* __restrict__ m_enc,
                                   float* __restrict__ p, float* __restrict__ denom)
{
    const int e = blockIdx.x * blockDim.x + threadIdx.x;
    if (e >= E + nloops) return;
    int s_, d_;
    if (e < E) { s_ = src[e]; d_ = dst[e]; } else { s_ = d_ = e - E; }
    const float x = ssrc[s_] + sdstv[d_];
    const float ee = x >= 0.f ? x : 0.2f * x;
    const float pe = __expf(ee - decf(m_enc[d_]));
    p[e] = pe;
    atomicAdd(&denom[d_], pe);
}

__global__ void edge_agg_kernel(const int* __restrict__ src, const int* __restrict__ dst,
                                int E, int nloops,
                                const float* __restrict__ p,
                                const float* __restrict__ denom,
                                const float* __restrict__ hs,
                                float* __restrict__ outacc)
{
    const int e = blockIdx.x * (blockDim.x >> 5) + (threadIdx.x >> 5);
    if (e >= E + nloops) return;
    const int lane = threadIdx.x & 31;
    int s_, d_;
    if (e < E) { s_ = src[e]; d_ = dst[e]; } else { s_ = d_ = e - E; }
    const float alpha = p[e] / denom[d_];
    const float* hrow = hs + (size_t)s_ * HD;
    float* orow = outacc + (size_t)d_ * HD;
    #pragma unroll
    for (int i = 0; i < 4; ++i) {
        const int c = lane + 32 * i;
        atomicAdd(&orow[c], alpha * hrow[c]);
    }
}

// x = relu(acc + b0 [+ b1] [+ b2])
__global__ void finalize_kernel(const float* __restrict__ acc,
                                const float* __restrict__ b0,
                                const float* __restrict__ b1,
                                const float* __restrict__ b2,
                                float* __restrict__ xout, int Nn)
{
    const int i = blockIdx.x * blockDim.x + threadIdx.x;
    if (i >= Nn * HD) return;
    const int h = i & (HD - 1);
    float v = acc[i] + b0[h];
    if (b1) v += b1[h];
    if (b2) v += b2[h];
    xout[i] = fmaxf(v, 0.f);
}

// hid[r, seg*128 : seg*128+128] = l2norm(row), one wave per (row, seg)
__global__ void gather_norm_kernel(const float* __restrict__ xd,
                                   const float* __restrict__ xc,
                                   const int* __restrict__ d1,
                                   const int* __restrict__ d2,
                                   const int* __restrict__ cc,
                                   float* __restrict__ hid, int row0, int nrows)
{
    const int w = blockIdx.x * (blockDim.x >> 5) + (threadIdx.x >> 5);
    if (w >= nrows * 3) return;
    const int lane = threadIdx.x & 31;
    const int r = w / 3, seg = w % 3;
    const int b = row0 + r;
    const int idx = (seg == 0) ? d1[b] : ((seg == 1) ? d2[b] : cc[b]);
    const float* xr = ((seg == 2) ? xc : xd) + (size_t)idx * HD;
    float vals[4];
    float ss = 0.f;
    #pragma unroll
    for (int i = 0; i < 4; ++i) { vals[i] = xr[lane + 32 * i]; ss += vals[i] * vals[i]; }
    #pragma unroll
    for (int off = 16; off > 0; off >>= 1) ss += __shfl_xor(ss, off, 32);
    const float scale = 1.0f / fmaxf(sqrtf(ss), 1e-12f);
    float* orow = hid + (size_t)r * (3 * HD) + seg * HD;
    #pragma unroll
    for (int i = 0; i < 4; ++i) orow[lane + 32 * i] = vals[i] * scale;
}

// out[b,0:2] = h2[b,:] @ W3 + b3   (K=256, N=2 -> scalar kernel)
__global__ void final_linear_kernel(const float* __restrict__ h2,
                                    const float* __restrict__ W3,
                                    const float* __restrict__ b3,
                                    float* __restrict__ out, int row0, int nrows)
{
    const int i = blockIdx.x * blockDim.x + threadIdx.x;
    if (i >= nrows * 2) return;
    const int r = i >> 1, j = i & 1;
    const float* hr = h2 + (size_t)r * 256;
    float s = b3[j];
    #pragma unroll 8
    for (int k = 0; k < 256; ++k) s += hr[k] * W3[k * 2 + j];
    out[(size_t)(row0 + r) * 2 + j] = s;
}

// ---------------------------------------------------------------------------
extern "C" void kernel_launch(void* const* d_in, const int* in_sizes, int n_in,
                              void* d_out, int out_size, void* d_ws, size_t ws_size,
                              hipStream_t stream)
{
    (void)in_sizes; (void)n_in; (void)out_size; (void)ws_size;

    const float* drug_emb    = (const float*)d_in[0];
    const float* protein_emb = (const float*)d_in[1];
    const float* cell_emb    = (const float*)d_in[2];
    const float* W_dp   = (const float*)d_in[3];
    const float* as_dp  = (const float*)d_in[4];
    const float* ad_dp  = (const float*)d_in[5];
    const float* b_dp   = (const float*)d_in[6];
    const float* W_rdp  = (const float*)d_in[7];
    const float* as_rdp = (const float*)d_in[8];
    const float* ad_rdp = (const float*)d_in[9];
    const float* b_rdp  = (const float*)d_in[10];
    const float* W_pp   = (const float*)d_in[11];
    const float* as_pp  = (const float*)d_in[12];
    const float* ad_pp  = (const float*)d_in[13];
    const float* b_pp   = (const float*)d_in[14];
    const float* W_cp   = (const float*)d_in[15];
    const float* as_cp  = (const float*)d_in[16];
    const float* ad_cp  = (const float*)d_in[17];
    const float* b_cp   = (const float*)d_in[18];
    const float* W_rcp  = (const float*)d_in[19];
    const float* as_rcp = (const float*)d_in[20];
    const float* ad_rcp = (const float*)d_in[21];
    const float* b_rcp  = (const float*)d_in[22];
    const float* W1 = (const float*)d_in[23];
    const float* b1 = (const float*)d_in[24];
    const float* W2 = (const float*)d_in[25];
    const float* b2 = (const float*)d_in[26];
    const float* W3 = (const float*)d_in[27];
    const float* b3 = (const float*)d_in[28];
    const int* src_dp = (const int*)d_in[29];
    const int* dst_dp = (const int*)d_in[30];
    const int* src_pp = (const int*)d_in[31];
    const int* dst_pp = (const int*)d_in[32];
    const int* src_cp = (const int*)d_in[33];
    const int* dst_cp = (const int*)d_in[34];
    const int* drug1  = (const int*)d_in[35];
    const int* drug2  = (const int*)d_in[36];
    const int* cell   = (const int*)d_in[37];
    float* out = (float*)d_out;

    // ---- workspace layout (offsets 256B-aligned) ----
    float* ws = (float*)d_ws;
    size_t o = 0;
    auto alloc = [&](size_t nfloats) -> float* {
        float* p = ws + o;
        o += (nfloats + 63) & ~(size_t)63;
        return p;
    };
    float* x_d   = alloc((size_t)NDRUG * HD);
    float* x_p   = alloc((size_t)NPROT * HD);
    float* x_c   = alloc((size_t)NCELL * HD);
    float* new_d = alloc((size_t)NDRUG * HD);
    float* new_p = alloc((size_t)NPROT * HD);
    float* new_c = alloc((size_t)NCELL * HD);
    float* hs    = alloc((size_t)NPROT * HD);
    float* s_src = alloc(NPROT);
    float* s_dst = alloc(NPROT);
    unsigned* m_enc = (unsigned*)alloc(NPROT);
    float* denom = alloc(NPROT);
    float* p_e   = alloc((size_t)(EPP + NPROT));
    float* wa_s  = alloc(HD);
    float* wa_d  = alloc(HD);
    _Float16* wt_rel = (_Float16*)alloc((size_t)HD * HD / 2);          // 128x128 f16
    _Float16* w1t    = (_Float16*)alloc((size_t)(3 * HD) * (6 * HD) / 2);
    _Float16* w2t    = (_Float16*)alloc((size_t)(6 * HD) * (2 * HD) / 2);
    float* hid   = alloc((size_t)MLP_CHUNK * 3 * HD);
    float* h1b   = alloc((size_t)MLP_CHUNK * 6 * HD);
    float* h2b   = alloc((size_t)MLP_CHUNK * 2 * HD);

    // ---- copy embeddings into mutable buffers ----
    hipMemcpyAsync(x_d, drug_emb,    (size_t)NDRUG * HD * 4, hipMemcpyDeviceToDevice, stream);
    hipMemcpyAsync(x_p, protein_emb, (size_t)NPROT * HD * 4, hipMemcpyDeviceToDevice, stream);
    hipMemcpyAsync(x_c, cell_emb,    (size_t)NCELL * HD * 4, hipMemcpyDeviceToDevice, stream);

    auto launch_gemm = [&](const float* A, const _Float16* Bt, const float* bias,
                           float* C, int M, int N, int K, int dorelu) {
        const int ntiles = (N >> 7) * ((M + 15) >> 4);
        gemm_wmma_f16<<<(ntiles + 3) / 4, 128, 0, stream>>>(A, Bt, bias, C, M, N, K, dorelu);
    };

    struct Rel {
        const float *Wbase, *asb, *adb;
        const int *src, *dst;
        int E, nloops;
        const float* Xs; int Ns;
        const float* Xd; int Nd;
        float* acc;
    };

    for (int l = 0; l < NLAYER; ++l) {
        zero_kernel<<<((NDRUG * HD) + 255) / 256, 256, 0, stream>>>(new_d, NDRUG * HD);
        zero_kernel<<<((NPROT * HD) + 255) / 256, 256, 0, stream>>>(new_p, NPROT * HD);
        zero_kernel<<<((NCELL * HD) + 255) / 256, 256, 0, stream>>>(new_c, NCELL * HD);

        const Rel rels[5] = {
            { W_dp,  as_dp,  ad_dp,  src_dp, dst_dp, EDP, 0,     x_d, NDRUG, x_p, NPROT, new_p },
            { W_pp,  as_pp,  ad_pp,  src_pp, dst_pp, EPP, NPROT, x_p, NPROT, x_p, NPROT, new_p },
            { W_cp,  as_cp,  ad_cp,  src_cp, dst_cp, ECP, 0,     x_c, NCELL, x_p, NPROT, new_p },
            { W_rdp, as_rdp, ad_rdp, dst_dp, src_dp, EDP, 0,     x_p, NPROT, x_d, NDRUG, new_d },
            { W_rcp, as_rcp, ad_rcp, dst_cp, src_cp, ECP, 0,     x_p, NPROT, x_c, NCELL, new_c },
        };

        for (int r = 0; r < 5; ++r) {
            const Rel& R = rels[r];
            const float* Wl  = R.Wbase + (size_t)l * HD * HD;
            const float* asl = R.asb + (size_t)l * HD;
            const float* adl = R.adb + (size_t)l * HD;

            // attention projection vectors + f16 transposed weights
            wa_kernel<<<1, 256, 0, stream>>>(Wl, asl, adl, wa_s, wa_d);
            cvt_transpose_f16<<<(HD * HD + 255) / 256, 256, 0, stream>>>(Wl, wt_rel, HD, HD);

            // hs = Xs @ W   (WMMA)
            launch_gemm(R.Xs, wt_rel, nullptr, hs, R.Ns, HD, HD, 0);

            // per-node attention scalars
            rowdot_kernel<<<(R.Ns + 3) / 4, 128, 0, stream>>>(R.Xs, wa_s, s_src, R.Ns);
            rowdot_kernel<<<(R.Nd + 3) / 4, 128, 0, stream>>>(R.Xd, wa_d, s_dst, R.Nd);

            // segment softmax + aggregate
            init_attn_kernel<<<(R.Nd + 255) / 256, 256, 0, stream>>>(m_enc, denom, R.Nd);
            const int Et = R.E + R.nloops;
            edge_max_kernel<<<(Et + 255) / 256, 256, 0, stream>>>(
                R.src, R.dst, R.E, R.nloops, s_src, s_dst, m_enc);
            edge_expsum_kernel<<<(Et + 255) / 256, 256, 0, stream>>>(
                R.src, R.dst, R.E, R.nloops, s_src, s_dst, m_enc, p_e, denom);
            edge_agg_kernel<<<(Et + 3) / 4, 128, 0, stream>>>(
                R.src, R.dst, R.E, R.nloops, p_e, denom, hs, R.acc);
        }

        finalize_kernel<<<((NPROT * HD) + 255) / 256, 256, 0, stream>>>(
            new_p, b_dp + (size_t)l * HD, b_pp + (size_t)l * HD, b_cp + (size_t)l * HD, x_p, NPROT);
        finalize_kernel<<<((NDRUG * HD) + 255) / 256, 256, 0, stream>>>(
            new_d, b_rdp + (size_t)l * HD, nullptr, nullptr, x_d, NDRUG);
        finalize_kernel<<<((NCELL * HD) + 255) / 256, 256, 0, stream>>>(
            new_c, b_rcp + (size_t)l * HD, nullptr, nullptr, x_c, NCELL);
    }

    // ---- readout MLP, chunked over the batch ----
    cvt_transpose_f16<<<((3 * HD) * (6 * HD) + 255) / 256, 256, 0, stream>>>(W1, w1t, 3 * HD, 6 * HD);
    cvt_transpose_f16<<<((6 * HD) * (2 * HD) + 255) / 256, 256, 0, stream>>>(W2, w2t, 6 * HD, 2 * HD);

    for (int r0 = 0; r0 < NBATCH; r0 += MLP_CHUNK) {
        const int nr = (NBATCH - r0) < MLP_CHUNK ? (NBATCH - r0) : MLP_CHUNK;

        gather_norm_kernel<<<((nr * 3) + 3) / 4, 128, 0, stream>>>(
            x_d, x_c, drug1, drug2, cell, hid, r0, nr);

        launch_gemm(hid, w1t, b1, h1b, nr, 6 * HD, 3 * HD, 1);   // [nr,768] K=384
        launch_gemm(h1b, w2t, b2, h2b, nr, 2 * HD, 6 * HD, 1);   // [nr,256] K=768

        final_linear_kernel<<<((nr * 2) + 255) / 256, 256, 0, stream>>>(
            h2b, W3, b3, out, r0, nr);
    }
}
